// Encoder_53661321396262
// MI455X (gfx1250) — compile-verified, matrix operand order checked by reading, not run on hardware
//
#include <hip/hip_runtime.h>
#include <hip/hip_bf16.h>

// ---------------------------------------------------------------------------
// GRU encoder for MI455X (gfx1250, wave32, WMMA)
//   VOCAB=32000, H=1024, OUT=1024, SEQ=2048
//   Phase A: gi = emb[tok] @ w_ih^T + b_ih  via v_wmma_f32_16x16x32_bf16,
//            software-pipelined (ping-pong fragment buffers)
//   Phase B: persistent 16-block scan; h staged to LDS with
//            global_load_async_to_lds_b64 (ASYNCcnt path); one atomic grid
//            barrier per timestep; w_hh L2-resident fp32
//   Phase C: mean/std head mat-vecs
// ---------------------------------------------------------------------------

#define H        1024
#define H3       3072
#define SEQ      2048
#define SCAN_WG  16          // persistent blocks for the recurrence
#define SCAN_TPB 512         // 16 waves of 32

typedef __attribute__((ext_vector_type(16))) __bf16 v16bf;
typedef __attribute__((ext_vector_type(8)))  __bf16 v8bf;
typedef __attribute__((ext_vector_type(8)))  float  v8f;

// fp32 -> bf16 with round-to-nearest-even
__device__ __forceinline__ __bf16 f2bf(float f) {
    unsigned u = __float_as_uint(f);
    u += 0x7FFFu + ((u >> 16) & 1u);
    unsigned short hs = (unsigned short)(u >> 16);
    return __builtin_bit_cast(__bf16, hs);
}

// Load a 16x32 bf16 WMMA fragment from a row-major matrix.
// Per ISA 7.12.2 (16-bit A 16x32): lane L<16 -> row L, K in {k0+0..7, k0+16..23};
// lane L>=16 -> row L-16, K in {k0+8..15, k0+24..31}.  Two 16B loads per lane.
__device__ __forceinline__ v16bf load_frag_bf16(const __bf16* p, int ld, int k0) {
    const int lane = threadIdx.x & 31;
    const __bf16* row = p + (size_t)(lane & 15) * ld + k0 + ((lane >> 4) << 3);
    v8bf lo = *(const v8bf*)(row);
    v8bf hi = *(const v8bf*)(row + 16);
    v16bf r;
#pragma unroll
    for (int i = 0; i < 8; ++i) { r[i] = lo[i]; r[i + 8] = hi[i]; }
    return r;
}

__device__ __forceinline__ v8f wmma_bf16(v16bf a, v16bf b, v8f c) {
    return __builtin_amdgcn_wmma_f32_16x16x32_bf16(false, a, false, b,
                                                   (short)0, c, false, false);
}

// ---------------------------------------------------------------------------
// K0: copy h0, reset grid-barrier state (re-run every launch -> deterministic)
// ---------------------------------------------------------------------------
__global__ void init_state(const float* __restrict__ hidden,
                           float* __restrict__ hbuf, unsigned* __restrict__ bar) {
    int i = blockIdx.x * blockDim.x + threadIdx.x;
    if (i < H) hbuf[i] = hidden[i];
    if (i == 0) { bar[0] = 0u; bar[1] = 0u; }
}

// ---------------------------------------------------------------------------
// K1a: gather x = emb[tokens] as bf16 rows
// ---------------------------------------------------------------------------
__global__ __launch_bounds__(256) void gather_x(const int* __restrict__ tok,
                                                const float* __restrict__ emb,
                                                __bf16* __restrict__ xb) {
    const int s = blockIdx.x;
    const float* src = emb + (size_t)tok[s] * H;
    __bf16* dst = xb + (size_t)s * H;
    for (int c = threadIdx.x; c < H; c += 256) dst[c] = f2bf(src[c]);
}

// K1b: generic fp32 -> bf16 (w_ih)
__global__ __launch_bounds__(256) void cvt_bf16(const float* __restrict__ src,
                                                __bf16* __restrict__ dst, int n) {
    int i = blockIdx.x * blockDim.x + threadIdx.x;
    if (i < n) dst[i] = f2bf(src[i]);
}

// ---------------------------------------------------------------------------
// K2: gi[SEQ][3H] = x @ w_ih^T + b_ih  (bf16 WMMA, fp32 accumulate)
// Block = 256 threads (8 waves) -> 32(M) x 256(N) tile; wave -> 16 x 64 strip.
// K-loop is ping-pong pipelined: loads for slab B are issued before the WMMAs
// of slab A so the scheduler can overlap VMEM latency with matrix math.
// Grid = (SEQ/32, 3H/256) = (64, 12).
// ---------------------------------------------------------------------------
__global__ __launch_bounds__(256, 2) void gemm_gi(const __bf16* __restrict__ xb,
                                                  const __bf16* __restrict__ wb,
                                                  const float*  __restrict__ b_ih,
                                                  float* __restrict__ gi) {
    const int wave = threadIdx.x >> 5;
    const int lane = threadIdx.x & 31;
    const int m0 = blockIdx.x * 32 + (wave >> 2) * 16;
    const int n0 = blockIdx.y * 256 + (wave & 3) * 64;

    const __bf16* arows = xb + (size_t)m0 * H;
    const __bf16* w0 = wb + (size_t)(n0)      * H;
    const __bf16* w1 = wb + (size_t)(n0 + 16) * H;
    const __bf16* w2 = wb + (size_t)(n0 + 32) * H;
    const __bf16* w3 = wb + (size_t)(n0 + 48) * H;

    v8f acc[4] = {};

    // prologue: fill buffer A with slab k=0
    v16bf aA  = load_frag_bf16(arows, H, 0);
    v16bf bA0 = load_frag_bf16(w0, H, 0);
    v16bf bA1 = load_frag_bf16(w1, H, 0);
    v16bf bA2 = load_frag_bf16(w2, H, 0);
    v16bf bA3 = load_frag_bf16(w3, H, 0);

#pragma unroll 1
    for (int k0 = 0; k0 < H; k0 += 64) {
        // issue loads for slab k0+32 (buffer B) ...
        v16bf aB  = load_frag_bf16(arows, H, k0 + 32);
        v16bf bB0 = load_frag_bf16(w0, H, k0 + 32);
        v16bf bB1 = load_frag_bf16(w1, H, k0 + 32);
        v16bf bB2 = load_frag_bf16(w2, H, k0 + 32);
        v16bf bB3 = load_frag_bf16(w3, H, k0 + 32);
        // ... while consuming buffer A
        acc[0] = wmma_bf16(aA, bA0, acc[0]);
        acc[1] = wmma_bf16(aA, bA1, acc[1]);
        acc[2] = wmma_bf16(aA, bA2, acc[2]);
        acc[3] = wmma_bf16(aA, bA3, acc[3]);

        const int k2 = k0 + 64;
        if (k2 < H) {            // uniform branch; EXEC untouched (WMMA-safe)
            aA  = load_frag_bf16(arows, H, k2);
            bA0 = load_frag_bf16(w0, H, k2);
            bA1 = load_frag_bf16(w1, H, k2);
            bA2 = load_frag_bf16(w2, H, k2);
            bA3 = load_frag_bf16(w3, H, k2);
        }
        acc[0] = wmma_bf16(aB, bB0, acc[0]);
        acc[1] = wmma_bf16(aB, bB1, acc[1]);
        acc[2] = wmma_bf16(aB, bB2, acc[2]);
        acc[3] = wmma_bf16(aB, bB3, acc[3]);
    }

    // C/D layout: N = lane&15; lanes>=16 hold M rows 8..15; VGPR e -> M row e.
    const int mBase = m0 + ((lane >> 4) << 3);
#pragma unroll
    for (int j = 0; j < 4; ++j) {
        const int n = n0 + j * 16 + (lane & 15);
        const float bias = b_ih[n];
#pragma unroll
        for (int e = 0; e < 8; ++e)
            gi[(size_t)(mBase + e) * H3 + n] = acc[j][e] + bias;
    }
}

// ---------------------------------------------------------------------------
// K3: persistent GRU scan. 16 blocks x 512 threads; block b owns h[64b..64b+63]
// and the 192 matching rows of w_hh (r/z/n gates). One grid barrier per step.
// h is staged to LDS with the CDNA5 async global->LDS path (ASYNCcnt).
// ---------------------------------------------------------------------------
__global__ __launch_bounds__(SCAN_TPB) void gru_scan(const float* __restrict__ gi,
                                                     const float* __restrict__ w_hh,
                                                     const float* __restrict__ b_hh,
                                                     float* __restrict__ hbuf,      // [2][H]
                                                     unsigned* __restrict__ bar) {  // [2]
    __shared__ float hs[H];
    __shared__ float ghs[192];

    const int tid  = threadIdx.x;
    const int lane = tid & 31;
    const int wave = tid >> 5;
    const int base = blockIdx.x * 64;          // h-chunk owned by this block

    // loop-invariant: biases for the 64 owned elements
    float bh_r = 0.f, bh_z = 0.f, bh_n = 0.f;
    if (tid < 64) {
        bh_r = b_hh[base + tid];
        bh_z = b_hh[base + tid + H];
        bh_n = b_hh[base + tid + 2 * H];
    }

    // per-thread LDS destination for the async h stage (8 bytes/thread = 4 KB)
    const unsigned lds_off =
        (unsigned)(unsigned long long)(uintptr_t)(&hs[2 * tid]);

    for (int t = 0; t < SEQ; ++t) {
        const float* hcur  = hbuf + (t & 1) * H;
        float*       hnext = hbuf + ((t + 1) & 1) * H;

        // stage h into LDS via async DMA (global_load_async_to_lds_b64)
        {
            unsigned long long src =
                (unsigned long long)(uintptr_t)(hcur + 2 * tid);
            asm volatile("global_load_async_to_lds_b64 %0, %1, off"
                         :: "v"(lds_off), "v"(src)
                         : "memory");
            asm volatile("s_wait_asynccnt 0" ::: "memory");
        }
        __syncthreads();

        const float4* hs4 = (const float4*)hs;

        // 192 rows / 16 waves = 12 rows per wave; 1024-dot per row
#pragma unroll 1
        for (int rr = wave * 12; rr < wave * 12 + 12; ++rr) {
            const int gate = rr / 64;          // 0=r, 1=z, 2=n
            const int il   = rr & 63;
            const float4* w4 =
                (const float4*)(w_hh + (size_t)(gate * H + base + il) * H);
            float sum = 0.f;
#pragma unroll
            for (int i = 0; i < 8; ++i) {
                float4 w = w4[lane + 32 * i];
                float4 h4 = hs4[lane + 32 * i];
                sum = fmaf(w.x, h4.x, sum);
                sum = fmaf(w.y, h4.y, sum);
                sum = fmaf(w.z, h4.z, sum);
                sum = fmaf(w.w, h4.w, sum);
            }
#pragma unroll
            for (int off = 16; off; off >>= 1) sum += __shfl_xor(sum, off, 32);
            if (lane == 0) ghs[rr] = sum;
        }

        // prefetch next step's gi slice while the dots drain
        if (tid < 64 && t + 1 < SEQ) {
            const float* gn = gi + (size_t)(t + 1) * H3 + base + tid;
            __builtin_prefetch(gn, 0, 1);
            __builtin_prefetch(gn + H, 0, 1);
            __builtin_prefetch(gn + 2 * H, 0, 1);
        }
        __syncthreads();

        // gates + state update for the 64 owned elements
        if (tid < 64) {
            const int i = base + tid;
            const float gr = ghs[tid]       + bh_r;
            const float gz = ghs[tid + 64]  + bh_z;
            const float gn = ghs[tid + 128] + bh_n;
            const float* git = gi + (size_t)t * H3;
            const float r = 1.f / (1.f + __expf(-(git[i]     + gr)));
            const float z = 1.f / (1.f + __expf(-(git[i + H] + gz)));
            const float n = tanhf(git[i + 2 * H] + r * gn);
            hnext[i] = (1.f - z) * n + z * hs[i];
        }
        __syncthreads();

        // device-wide sense-reversal barrier (bar[0]=arrive, bar[1]=generation)
        if (tid == 0) {
            __threadfence();
            unsigned g = atomicAdd(bar + 1, 0u);           // read generation
            if (atomicAdd(bar, 1u) == SCAN_WG - 1u) {
                atomicExch(bar, 0u);
                __threadfence();
                atomicAdd(bar + 1, 1u);                    // release
            } else {
                while (atomicAdd(bar + 1, 0u) == g)
                    __builtin_amdgcn_s_sleep(2);
            }
            __threadfence();
        }
        __syncthreads();
    }
    // final h lives in hbuf[SEQ & 1] == hbuf[0]
}

// ---------------------------------------------------------------------------
// K4: out_mean / out_std = h @ W^T + b. Wave-per-row; grid.y picks mean/std.
// ---------------------------------------------------------------------------
__global__ __launch_bounds__(256) void head_matvec(const float* __restrict__ h,
                                                   const float* __restrict__ w_mean,
                                                   const float* __restrict__ b_mean,
                                                   const float* __restrict__ w_std,
                                                   const float* __restrict__ b_std,
                                                   float* __restrict__ out) {
    const int wave = threadIdx.x >> 5;
    const int lane = threadIdx.x & 31;
    const int j = blockIdx.x * 8 + wave;
    const float* w = blockIdx.y ? w_std  : w_mean;
    const float* b = blockIdx.y ? b_std  : b_mean;
    const float4* w4 = (const float4*)(w + (size_t)j * H);
    const float4* h4 = (const float4*)h;
    float sum = 0.f;
#pragma unroll
    for (int i = 0; i < 8; ++i) {
        float4 wv = w4[lane + 32 * i];
        float4 hv = h4[lane + 32 * i];
        sum = fmaf(wv.x, hv.x, sum);
        sum = fmaf(wv.y, hv.y, sum);
        sum = fmaf(wv.z, hv.z, sum);
        sum = fmaf(wv.w, hv.w, sum);
    }
#pragma unroll
    for (int off = 16; off; off >>= 1) sum += __shfl_xor(sum, off, 32);
    if (lane == 0) out[blockIdx.y * H + j] = sum + b[j];
}

// ---------------------------------------------------------------------------
// Host-side launch
// ---------------------------------------------------------------------------
extern "C" void kernel_launch(void* const* d_in, const int* in_sizes, int n_in,
                              void* d_out, int out_size, void* d_ws, size_t ws_size,
                              hipStream_t stream) {
    const int*   tok    = (const int*)  d_in[0];
    const float* hidden = (const float*)d_in[1];
    const float* emb    = (const float*)d_in[2];
    const float* w_ih   = (const float*)d_in[3];
    const float* w_hh   = (const float*)d_in[4];
    const float* b_ih   = (const float*)d_in[5];
    const float* b_hh   = (const float*)d_in[6];
    const float* w_mean = (const float*)d_in[7];
    const float* b_mean = (const float*)d_in[8];
    const float* w_std  = (const float*)d_in[9];
    const float* b_std  = (const float*)d_in[10];
    float* out = (float*)d_out;

    // workspace layout (256B aligned)
    char* ws = (char*)d_ws;
    size_t off = 0;
    auto alloc = [&](size_t bytes) {
        void* p = ws + off;
        off += (bytes + 255) & ~(size_t)255;
        return p;
    };
    __bf16*   xb    = (__bf16*)  alloc((size_t)SEQ * H * 2);      //  4 MB
    __bf16*   wb    = (__bf16*)  alloc((size_t)H3 * H * 2);       //  6 MB
    float*    gi    = (float*)   alloc((size_t)SEQ * H3 * 4);     // 24 MB
    float*    hbuf  = (float*)   alloc((size_t)2 * H * 4);        //  8 KB
    unsigned* bar   = (unsigned*)alloc(2 * sizeof(unsigned));
    (void)ws_size; (void)n_in; (void)in_sizes; (void)out_size;

    // K0: init h0 + barrier state
    init_state<<<4, 256, 0, stream>>>(hidden, hbuf, bar);

    // K1: precision conversion for the WMMA GEMM
    gather_x<<<SEQ, 256, 0, stream>>>(tok, emb, xb);
    cvt_bf16<<<(H3 * H) / 256, 256, 0, stream>>>(w_ih, wb, H3 * H);

    // K2: gi = x @ w_ih^T + b_ih   (WMMA bf16 -> fp32, pipelined)
    gemm_gi<<<dim3(SEQ / 32, H3 / 256), 256, 0, stream>>>(xb, wb, b_ih, gi);

    // K3: persistent sequential scan (async h staging + grid barrier)
    gru_scan<<<SCAN_WG, SCAN_TPB, 0, stream>>>(gi, w_hh, b_hh, hbuf, bar);

    // K4: output heads (final h in hbuf[0])
    head_matvec<<<dim3(H / 8, 2), 256, 0, stream>>>(hbuf, w_mean, b_mean,
                                                    w_std, b_std, out);
}
